// MultiHeadAttention_38465727103751
// MI455X (gfx1250) — compile-verified, matrix operand order checked by reading, not run on hardware
//
#include <hip/hip_runtime.h>

// ---------------------------------------------------------------------------
// MI455X / gfx1250 multi-head attention forward (B=4, S=2048, D=768, H=12,
// Hd=64). Compute-bound (~90 GFLOP vs ~100MB traffic): all matmuls on
// v_wmma_f32_16x16x32_bf16 (fp32 accumulate), wave32.
// GEMM tiles staged by the Tensor Data Mover (tensor_load_to_lds, TENSORcnt),
// double-buffered in LDS; all LDS fragment reads are contiguous b128 loads.
// ---------------------------------------------------------------------------

#define BATCH 4
#define SEQ   2048
#define DIM   768
#define HEADS 12
#define HD    64
#define BS    (BATCH * SEQ)      // 8192
#define D3    (3 * DIM)          // 2304

typedef __attribute__((ext_vector_type(16))) __bf16          v16bf;
typedef __attribute__((ext_vector_type(8)))  float           v8f;
typedef __attribute__((ext_vector_type(16))) unsigned short  us16;
typedef __attribute__((ext_vector_type(8)))  unsigned short  us8;
typedef __attribute__((ext_vector_type(4)))  unsigned int    uint4v;
typedef __attribute__((ext_vector_type(8)))  int             int8v;
typedef __attribute__((ext_vector_type(4)))  int             int4v;

__device__ __forceinline__ unsigned short f2bf(float f) {
    union { float f; unsigned int u; } x; x.f = f;
    unsigned int u = x.u;
    unsigned int r = ((u >> 16) & 1u) + 0x7FFFu;   // round-to-nearest-even
    return (unsigned short)((u + r) >> 16);
}

__device__ __forceinline__ v8f wmma_bf16(us16 a, us16 b, v8f c) {
    return __builtin_amdgcn_wmma_f32_16x16x32_bf16(
        /*neg_a=*/false, __builtin_bit_cast(v16bf, a),
        /*neg_b=*/false, __builtin_bit_cast(v16bf, b),
        /*c_mod=*/(short)0, c, /*reuse_a=*/false, /*reuse_b=*/false);
}

// combine two contiguous 8-element LDS reads into one 16-element fragment
__device__ __forceinline__ us16 frag_from_lds(const unsigned short* p, int off_lo, int off_hi) {
    us8 lo = *(const us8*)(p + off_lo);
    us8 hi = *(const us8*)(p + off_hi);
    us16 f;
    for (int j = 0; j < 8; ++j) { f[j] = lo[j]; f[j + 8] = hi[j]; }
    return f;
}

// ---------------------------------------------------------------------------
// TDM: issue a 2D tile load (16-bit elements) into LDS via the Tensor Data
// Mover. Descriptor bit layout per CDNA5 ISA §8.3/8.4 (D# groups 0 and 1).
// All parameters are workgroup-uniform -> descriptor lives in SGPRs.
// ---------------------------------------------------------------------------
__device__ __forceinline__ void tdm_load_tile_2d(
    unsigned lds_addr,            // LDS byte address of tile destination
    const void* gaddr,            // global byte address of tile start
    unsigned tensor_d0,           // remaining row length (elements)
    unsigned tensor_d1,           // remaining rows
    unsigned tile_d0,             // tile width  (elements)
    unsigned tile_d1,             // tile height (rows)
    unsigned stride0)             // row stride (elements)
{
    unsigned long long ga = (unsigned long long)gaddr;
    uint4v g0;
    g0[0] = 1u;                                                  // count=1, user D#
    g0[1] = lds_addr;                                            // lds_addr[31:0]
    g0[2] = (unsigned)(ga & 0xffffffffu);                        // global_addr[31:0]
    g0[3] = (unsigned)((ga >> 32) & 0x1ffffffu) | (2u << 30);    // addr[56:32], type=2
    int8v g1;
    g1[0] = (int)(1u << 16);                                     // data_size=2B, wg_mask=0
    g1[1] = (int)((tensor_d0 & 0xffffu) << 16);                  // tensor_dim0[15:0]
    g1[2] = (int)((tensor_d0 >> 16) | ((tensor_d1 & 0xffffu) << 16));
    g1[3] = (int)((tensor_d1 >> 16) | (tile_d0 << 16));          // tile_dim0
    g1[4] = (int)(tile_d1 & 0xffffu);                            // tile_dim1, tile_dim2=0
    g1[5] = (int)stride0;                                        // dim0_stride[31:0]
    g1[6] = 0;
    g1[7] = 0;
    int4v z4 = (int4v)0;
#if __clang_major__ >= 23
    int8v z8 = (int8v)0;
    __builtin_amdgcn_tensor_load_to_lds(g0, g1, z4, z4, z8, 0);
#else
    __builtin_amdgcn_tensor_load_to_lds(g0, g1, z4, z4, 0);
#endif
}

__device__ __forceinline__ unsigned lds_addr_of(const void* p) {
    // generic pointers to LDS carry the LDS byte offset in the low 32 bits
    return (unsigned)(size_t)p;
}

// ---------------------------------------------------------------------------
// fp32 -> bf16 bulk convert (row-major copy)
// ---------------------------------------------------------------------------
__global__ void mha_cvt_bf16(const float* __restrict__ src,
                             unsigned short* __restrict__ dst, int n) {
    int i = blockIdx.x * blockDim.x + threadIdx.x;
    if (i < n) dst[i] = f2bf(src[i]);
}

// fp32 [K][N] -> bf16 transposed [N][K] (one-time weight prep)
__global__ void mha_cvt_bf16_t(const float* __restrict__ src,
                               unsigned short* __restrict__ dst, int K, int N) {
    int i = blockIdx.x * blockDim.x + threadIdx.x;
    if (i < K * N) {
        int n = i / K, k = i % K;
        dst[i] = f2bf(src[(size_t)k * N + n]);
    }
}

// ---------------------------------------------------------------------------
// Tiled bf16 GEMM: C[M,N] = A[M,K] @ Bt[N,K]^T + bias[N]
//   block tile 64x64, 128 threads (4 waves), wave computes 16(M) x 64(N).
//   A/B tiles are DMA'd by the TDM into double-buffered LDS: wave 0 issues
//   tensor_load_to_lds one K-step ahead, retires with s_wait_tensorcnt, and
//   the workgroup barrier publishes the tile. VALU/VMEM pipes stay free for
//   WMMA + fragment ds_load_b128.
//   mode 0: scatter bf16 into qkv buffer; q,k -> [.][B][H][S][Hd],
//           v -> [B][H][Hd][S] (transposed for the P@V WMMA feed)
//   mode 1: write fp32 row-major out[M,N]
// ---------------------------------------------------------------------------
__global__ __launch_bounds__(128)
void mha_gemm_bf16(const unsigned short* __restrict__ A,    // [M][K]
                   const unsigned short* __restrict__ Bt,   // [N][K]
                   const float* __restrict__ bias,
                   unsigned short* __restrict__ outQKV,
                   float* __restrict__ outF,
                   int M, int N, int K, int mode) {
    const int TK = 32;
    __shared__ unsigned short As [2][64][TK];   // TDM writes tiles densely
    __shared__ unsigned short Bts[2][64][TK];

    const int tid  = threadIdx.x;
    const int wave = tid >> 5;
    const int lane = tid & 31;
    const int tileN0 = blockIdx.x * 64;
    const int tileM0 = blockIdx.y * 64;

    v8f acc[4] = {};

    const int mrow = wave * 16 + (lane & 15);
    const int kb   = (lane >> 4) * 8;            // 0 or 8

    const int nsteps = K / TK;

    // prologue: DMA first K-step tiles into buffer 0
    if (wave == 0) {
        tdm_load_tile_2d(lds_addr_of(&As [0][0][0]), &A [(size_t)tileM0 * K],
                         K, M - tileM0, TK, 64, K);
        tdm_load_tile_2d(lds_addr_of(&Bts[0][0][0]), &Bt[(size_t)tileN0 * K],
                         K, N - tileN0, TK, 64, K);
    }

    for (int i = 0; i < nsteps; ++i) {
        const int buf = i & 1;
        if (i + 1 < nsteps) {
            const int k1 = (i + 1) * TK;
            if (wave == 0) {    // DMA next K-step into the other buffer
                tdm_load_tile_2d(lds_addr_of(&As [buf ^ 1][0][0]),
                                 &A [(size_t)tileM0 * K + k1],
                                 K - k1, M - tileM0, TK, 64, K);
                tdm_load_tile_2d(lds_addr_of(&Bts[buf ^ 1][0][0]),
                                 &Bt[(size_t)tileN0 * K + k1],
                                 K - k1, N - tileN0, TK, 64, K);
            }
            __builtin_amdgcn_s_wait_tensorcnt(2);   // tile i landed (in-order TDM)
        } else {
            __builtin_amdgcn_s_wait_tensorcnt(0);
        }
        __syncthreads();   // publish tile i to all waves

        us16 af = frag_from_lds(&As[buf][mrow][0], kb, kb + 16);
        for (int a = 0; a < 4; ++a) {
            const int col = a * 16 + (lane & 15);
            us16 bf = frag_from_lds(&Bts[buf][col][0], kb, kb + 16);
            acc[a] = wmma_bf16(af, bf, acc[a]);
        }
        __syncthreads();   // tile i consumed; its buffer may be overwritten
    }

    // epilogue: D layout -> M = wave*16 + 8*(lane>>4) + i, N = a*16 + (lane&15)
    const int rowE = (lane >> 4) * 8;
    const int colL = lane & 15;
    for (int a = 0; a < 4; ++a) {
        for (int i = 0; i < 8; ++i) {
            int r = tileM0 + wave * 16 + rowE + i;
            int c = tileN0 + a * 16 + colL;
            float v = acc[a][i] + bias[c];
            if (mode == 0) {
                int which = c / DIM;           // 0=q 1=k 2=v
                int dd = c % DIM;
                int h  = dd >> 6;
                int hd = dd & 63;
                int b  = r >> 11;              // r / SEQ
                int s  = r & 2047;
                size_t base = ((size_t)(which * BATCH + b) * HEADS + h) * (size_t)SEQ * HD;
                size_t idx  = (which == 2) ? base + (size_t)hd * SEQ + s     // V transposed
                                           : base + (size_t)s * HD + hd;     // Q,K row-major
                outQKV[idx] = f2bf(v);
            } else {
                outF[(size_t)r * N + c] = v;
            }
        }
    }
}

// ---------------------------------------------------------------------------
// Attention: one workgroup per (b, h, 16-query tile). 128 threads / 4 waves.
// 16x2048 fp32 score strip lives in LDS (128KB of the WGP's 320KB),
// softmax via wave32 shuffle reductions, P bf16 in LDS, O = P @ V via WMMA.
// ---------------------------------------------------------------------------
__global__ __launch_bounds__(128)
void mha_attn(const unsigned short* __restrict__ qkv,
              unsigned short* __restrict__ aout) {        // [B][S][D] bf16
    extern __shared__ char smem[];
    float*          sc  = (float*)smem;                                   // [16][2048]
    unsigned short* pm  = (unsigned short*)(smem + 16 * SEQ * 4);         // [16][2048]
    unsigned short* qs  = pm + 16 * SEQ;                                  // [16][64]
    unsigned short* kts = qs + 16 * HD;                                   // [64][64]
    unsigned short* vts = kts + 64 * HD;                                  // [64][32]  (V^T tile)
    float*          rowsum = (float*)(vts + 64 * 32);                     // [16]

    const int tid  = threadIdx.x;
    const int wave = tid >> 5;
    const int lane = tid & 31;

    const int bid = blockIdx.x;
    const int mt  = bid & 127;                  // query tile (S/16 = 128)
    const int h   = (bid >> 7) % HEADS;
    const int b   = bid / (128 * HEADS);
    const int s0  = mt * 16;

    const size_t headStride = (size_t)SEQ * HD;
    const unsigned short* qp = qkv + ((size_t)(0 * BATCH + b) * HEADS + h) * headStride;
    const unsigned short* kp = qkv + ((size_t)(1 * BATCH + b) * HEADS + h) * headStride;
    const unsigned short* vp = qkv + ((size_t)(2 * BATCH + b) * HEADS + h) * headStride; // [Hd][S]

    // ---- load Q tile [16][64] ----
    {
        int r = tid >> 3, c = (tid & 7) * 8;
        *(us8*)&qs[r * HD + c] = *(const us8*)&qp[(size_t)(s0 + r) * HD + c];
    }
    __syncthreads();

    const int mrow = lane & 15;
    const int kb   = (lane >> 4) * 8;

    // Q A-fragments for K-window [0,32) and [32,64)
    us16 aq0 = frag_from_lds(&qs[mrow * HD], kb, kb + 16);
    us16 aq1 = frag_from_lds(&qs[mrow * HD], 32 + kb, 32 + kb + 16);

    // ---- scores: S = (Q @ K^T) * 1/sqrt(64) streamed into LDS strip ----
    for (int kt0 = 0; kt0 < SEQ; kt0 += 64) {
        for (int it = 0; it < 4; ++it) {
            int idx = tid + it * 128;
            int r = idx >> 3, c = (idx & 7) * 8;
            *(us8*)&kts[r * HD + c] = *(const us8*)&kp[(size_t)(kt0 + r) * HD + c];
        }
        __syncthreads();

        // B frag of K^T: B[kk][n] = Ktile[n][kk] -> contiguous LDS reads
        const int keyl = wave * 16 + (lane & 15);
        us16 bk0 = frag_from_lds(&kts[keyl * HD], kb, kb + 16);
        us16 bk1 = frag_from_lds(&kts[keyl * HD], 32 + kb, 32 + kb + 16);

        v8f s_acc = {};
        s_acc = wmma_bf16(aq0, bk0, s_acc);
        s_acc = wmma_bf16(aq1, bk1, s_acc);

        const int rowE = (lane >> 4) * 8;
        const int colL = lane & 15;
        for (int i = 0; i < 8; ++i)
            sc[(rowE + i) * SEQ + kt0 + wave * 16 + colL] = s_acc[i] * 0.125f;
        __syncthreads();
    }

    // ---- softmax over 2048 per row: 8 lanes per row, shfl_xor reduce ----
    {
        const int srow = tid >> 3;
        const int sub  = tid & 7;
        float m = -1e30f;
        for (int j = sub; j < SEQ; j += 8) m = fmaxf(m, sc[srow * SEQ + j]);
        m = fmaxf(m, __shfl_xor(m, 1, 32));
        m = fmaxf(m, __shfl_xor(m, 2, 32));
        m = fmaxf(m, __shfl_xor(m, 4, 32));
        float lsum = 0.f;
        for (int j = sub; j < SEQ; j += 8) {
            float e = __expf(sc[srow * SEQ + j] - m);
            lsum += e;
            pm[srow * SEQ + j] = f2bf(e);       // unnormalized P; divide at end
        }
        lsum += __shfl_xor(lsum, 1, 32);
        lsum += __shfl_xor(lsum, 2, 32);
        lsum += __shfl_xor(lsum, 4, 32);
        if (sub == 0) rowsum[srow] = lsum;
    }
    __syncthreads();

    // ---- O = P @ V : each wave owns a 16-wide Hd slice ----
    v8f o = {};
    const int nh = wave * 16;
    for (int ks = 0; ks < SEQ; ks += 32) {
        // stage V^T tile [64 hd][32 keys], pure b128 traffic
        for (int it = 0; it < 2; ++it) {
            int idx = tid + it * 128;
            int r = idx >> 2, c = (idx & 3) * 8;
            *(us8*)&vts[r * 32 + c] = *(const us8*)&vp[(size_t)r * SEQ + ks + c];
        }
        __syncthreads();

        // A frag from P (contiguous), B frag from V^T (contiguous)
        us16 ap = frag_from_lds(&pm[mrow * SEQ + ks], kb, kb + 16);
        us16 bv = frag_from_lds(&vts[(nh + (lane & 15)) * 32], kb, kb + 16);
        o = wmma_bf16(ap, bv, o);
        __syncthreads();
    }

    // ---- normalize + scatter to [B][S][D] bf16 ----
    const int rowE = (lane >> 4) * 8;
    const int colL = lane & 15;
    for (int i = 0; i < 8; ++i) {
        int r = rowE + i;
        float v = o[i] / rowsum[r];
        int s = s0 + r;
        aout[((size_t)b * SEQ + s) * DIM + h * HD + nh + colL] = f2bf(v);
    }
}

// ---------------------------------------------------------------------------
// launch
// ---------------------------------------------------------------------------
extern "C" void kernel_launch(void* const* d_in, const int* in_sizes, int n_in,
                              void* d_out, int out_size, void* d_ws, size_t ws_size,
                              hipStream_t stream) {
    (void)in_sizes; (void)n_in; (void)out_size; (void)ws_size;
    const float* x      = (const float*)d_in[0];
    const float* w_qkv  = (const float*)d_in[1];
    const float* b_qkv  = (const float*)d_in[2];
    const float* w_proj = (const float*)d_in[3];
    const float* b_proj = (const float*)d_in[4];
    float* out = (float*)d_out;

    // workspace carve-up (bf16 as ushort):
    unsigned short* xb     = (unsigned short*)d_ws;        // [BS][DIM]
    unsigned short* wqkvt  = xb + (size_t)BS * DIM;        // [3D][DIM]   (W^T)
    unsigned short* wprojt = wqkvt + (size_t)DIM * D3;     // [DIM][DIM]  (W^T)
    unsigned short* qkvb   = wprojt + (size_t)DIM * DIM;   // q,k:[.][B][H][S][Hd]; v:[B][H][Hd][S]
    unsigned short* aoutb  = qkvb + (size_t)3 * BS * DIM;  // [B][S][DIM]

    const int nx = BS * DIM, nwq = DIM * D3, nwp = DIM * DIM;
    mha_cvt_bf16  <<<(nx  + 255) / 256, 256, 0, stream>>>(x, xb, nx);
    mha_cvt_bf16_t<<<(nwq + 255) / 256, 256, 0, stream>>>(w_qkv,  wqkvt,  DIM, D3);
    mha_cvt_bf16_t<<<(nwp + 255) / 256, 256, 0, stream>>>(w_proj, wprojt, DIM, DIM);

    // QKV projection: [8192,768] @ [768,2304] -> scattered bf16 q/k/v (v transposed)
    mha_gemm_bf16<<<dim3(D3 / 64, BS / 64), 128, 0, stream>>>(
        xb, wqkvt, b_qkv, qkvb, nullptr, BS, D3, DIM, 0);

    // attention per (b,h,16-row tile)
    size_t attn_lds = (size_t)16 * SEQ * 4 + (size_t)16 * SEQ * 2 +
                      (size_t)(16 * HD + 64 * HD + 64 * 32) * 2 + 16 * 4;
    mha_attn<<<BATCH * HEADS * (SEQ / 16), 128, attn_lds, stream>>>(qkvb, aoutb);

    // output projection: [8192,768] @ [768,768] + bias -> fp32 out
    mha_gemm_bf16<<<dim3(DIM / 64, BS / 64), 128, 0, stream>>>(
        aoutb, wprojt, b_proj, nullptr, out, BS, DIM, DIM, 1);
}
